// GNN_75110388073003
// MI455X (gfx1250) — matrix operand order repaired and checked by reading, not compiled, not run on hardware
//
#include <hip/hip_runtime.h>
#include <stdint.h>

// Problem constants (match reference): B*N = 32768 nodes, DIM=128, 3 layers, E=524288
#define NNODES 32768
#define DIM    128
#define NEDGE  524288
#define NLAYER 3

typedef __attribute__((ext_vector_type(16))) __bf16 v16bf;
typedef __attribute__((ext_vector_type(2)))  __bf16 v2bf;
typedef __attribute__((ext_vector_type(8)))  float  v8f;

union BF16x16 { unsigned u[8]; v16bf v; };

// pack two f32 -> two bf16 in one dword (hardware cvt if available, else RNE bit-trick)
#if __has_builtin(__builtin_amdgcn_cvt_pk_bf16_f32)
__device__ __forceinline__ unsigned pack2bf(float a, float b) {
    v2bf r = __builtin_amdgcn_cvt_pk_bf16_f32(a, b);
    return __builtin_bit_cast(unsigned, r);
}
#else
__device__ __forceinline__ unsigned pack2bf(float a, float b) {
    union { float f; unsigned u; } ua, ub;
    ua.f = a; ub.f = b;
    unsigned ra = ua.u + 0x7FFFu + ((ua.u >> 16) & 1u);
    unsigned rb = ub.u + 0x7FFFu + ((ub.u >> 16) & 1u);
    return (ra >> 16) | (rb & 0xFFFF0000u);
}
#endif

// ---------------- degree / normalization ----------------
__global__ void deg_init(float* __restrict__ deg) {
    int i = blockIdx.x * blockDim.x + threadIdx.x;
    if (i < NNODES) deg[i] = 1.0f;              // self-loop contribution
}

__global__ void deg_acc(const int* __restrict__ cols, float* __restrict__ deg) {
    int e = blockIdx.x * blockDim.x + threadIdx.x;
    if (e < NEDGE) unsafeAtomicAdd(&deg[cols[e]], 1.0f);
}

__global__ void deg_rsqrt(float* __restrict__ deg) {
    int i = blockIdx.x * blockDim.x + threadIdx.x;
    if (i < NNODES) deg[i] = rsqrtf(deg[i]);    // deg >= 1 always
}

// ---------------- GEMM: hW = h @ W  (bf16 WMMA, f32 accumulate) ----------------
// Block: 256 threads = 8 waves; each wave computes 16 rows x 128 cols.
// Grid: NNODES/128 blocks. W staged in LDS as packed bf16 row-pairs: lds[k/2][n].
__global__ __launch_bounds__(256) void gemm_bf16_wmma(
    const float* __restrict__ h, const float* __restrict__ W,
    float* __restrict__ hW)
{
    __shared__ unsigned ldsW[64 * 128];  // 32 KB: {W[2k2][n], W[2k2+1][n]} packed
    const int tid = threadIdx.x;
    for (int p = tid; p < 64 * 128; p += 256) {
        int k2 = p >> 7;
        int n  = p & 127;
        ldsW[p] = pack2bf(W[(2 * k2) * DIM + n], W[(2 * k2 + 1) * DIM + n]);
    }
    __syncthreads();

    const int wave = tid >> 5;
    const int lane = tid & 31;
    const int m16  = lane & 15;
    const int hf   = lane >> 4;                // lane half (K-block select)
    const int rowbase = blockIdx.x * 128 + wave * 16;
    const int m = rowbase + m16;

    // A fragments: 16x32 bf16 per K-chunk, ISA lane map:
    // vgpr v(0..3): K = 2v,2v+1 (+8 for upper half); v(4..7): +16
    BF16x16 afrag[4];
#pragma unroll
    for (int kc = 0; kc < 4; ++kc) {
#pragma unroll
        for (int v = 0; v < 8; ++v) {
            int k = kc * 32 + ((v & 4) ? 16 : 0) + hf * 8 + 2 * (v & 3);
            const float* p = h + m * DIM + k;
            afrag[kc].u[v] = pack2bf(p[0], p[1]);
        }
    }

#pragma unroll
    for (int nt = 0; nt < 8; ++nt) {
        v8f acc = {};
#pragma unroll
        for (int kc = 0; kc < 4; ++kc) {
            // B fragment 32x16: vgpr v holds rows k = kc*32 + ((v&4)?16:0) + hf*8 + 2(v&3), +1
            BF16x16 bfrag;
#pragma unroll
            for (int v = 0; v < 8; ++v)
                bfrag.u[v] = ldsW[(kc * 16 + ((v & 4) ? 8 : 0) + hf * 4 + (v & 3)) * 128 + nt * 16 + m16];
            acc = __builtin_amdgcn_wmma_f32_16x16x32_bf16(
                false, afrag[kc].v, false, bfrag.v, (short)0, acc, false, false);
        }
        // C/D layout: vgpr r -> M = r (lanes 0-15) / r+8 (lanes 16-31), N = lane%16
#pragma unroll
        for (int r = 0; r < 8; ++r)
            hW[(rowbase + r + hf * 8) * DIM + nt * 16 + m16] = acc[r];
    }
}

// ---------------- edge scatter: agg[col] += hW[row] * dis[row]*dis[col] ----------------
// One wave per edge: coalesced 512B gather, 4 f32 atomics per lane into L2.
__global__ __launch_bounds__(256) void edge_scatter(
    const int* __restrict__ rows, const int* __restrict__ cols,
    const float* __restrict__ dis, const float* __restrict__ hW,
    float* __restrict__ agg)
{
    int wid  = (blockIdx.x * blockDim.x + threadIdx.x) >> 5;
    int lane = threadIdx.x & 31;
    if (wid >= NEDGE) return;
    int r = rows[wid];
    int c = cols[wid];
    float w = dis[r] * dis[c];
    float4 v = ((const float4*)(hW + r * DIM))[lane];
    float* dst = agg + c * DIM + lane * 4;
    unsafeAtomicAdd(dst + 0, v.x * w);
    unsafeAtomicAdd(dst + 1, v.y * w);
    unsafeAtomicAdd(dst + 2, v.z * w);
    unsafeAtomicAdd(dst + 3, v.w * w);
}

// ---------------- finalize: h = relu(h + agg + hW*dis^2 + b), float4-wide ----------------
__global__ void finalize(const float* __restrict__ agg, const float* __restrict__ hW,
                         const float* __restrict__ dis, const float* __restrict__ bias,
                         float* __restrict__ hio)
{
    int idx = blockIdx.x * blockDim.x + threadIdx.x;   // NNODES*32 float4 lanes
    if (idx >= NNODES * (DIM / 4)) return;
    int i  = idx >> 5;          // node
    int c4 = idx & 31;          // float4 column
    float d  = dis[i];
    float dd = d * d;
    float4 hv = ((float4*)hio)[idx];
    float4 av = ((const float4*)agg)[idx];
    float4 wv = ((const float4*)hW)[idx];
    float4 bv = ((const float4*)bias)[c4];
    float4 o;
    o.x = fmaxf(hv.x + av.x + wv.x * dd + bv.x, 0.0f);
    o.y = fmaxf(hv.y + av.y + wv.y * dd + bv.y, 0.0f);
    o.z = fmaxf(hv.z + av.z + wv.z * dd + bv.z, 0.0f);
    o.w = fmaxf(hv.w + av.w + wv.w * dd + bv.w, 0.0f);
    ((float4*)hio)[idx] = o;
}

extern "C" void kernel_launch(void* const* d_in, const int* in_sizes, int n_in,
                              void* d_out, int out_size, void* d_ws, size_t ws_size,
                              hipStream_t stream) {
    (void)in_sizes; (void)n_in; (void)out_size; (void)ws_size;
    const float* x  = (const float*)d_in[0];        // [B,N,DIM] f32
    const int*   ei = (const int*)d_in[1];          // [2,E] i32
    const float* Ws = (const float*)d_in[2];        // [NL,DIM,DIM] f32
    const float* bs = (const float*)d_in[3];        // [NL,DIM] f32
    float* out = (float*)d_out;                     // h buffer across layers
    float* ws  = (float*)d_ws;

    float* dis = ws;                                // NNODES (deg -> rsqrt in place)
    float* hW  = ws + NNODES;                       // NNODES*DIM
    float* agg = hW + (size_t)NNODES * DIM;         // NNODES*DIM

    const int* rows = ei;
    const int* cols = ei + NEDGE;

    deg_init <<<NNODES / 256, 256, 0, stream>>>(dis);
    deg_acc  <<<NEDGE  / 256, 256, 0, stream>>>(cols, dis);
    deg_rsqrt<<<NNODES / 256, 256, 0, stream>>>(dis);

    hipMemcpyAsync(out, x, sizeof(float) * NNODES * DIM,
                   hipMemcpyDeviceToDevice, stream);

    for (int l = 0; l < NLAYER; ++l) {
        gemm_bf16_wmma<<<NNODES / 128, 256, 0, stream>>>(out, Ws + l * DIM * DIM, hW);
        hipMemsetAsync(agg, 0, sizeof(float) * NNODES * DIM, stream);
        edge_scatter<<<(NEDGE * 32) / 256, 256, 0, stream>>>(rows, cols, dis, hW, agg);
        finalize<<<(NNODES * (DIM / 4)) / 256, 256, 0, stream>>>(agg, hW, dis,
                                                                 bs + l * DIM, out);
    }
}
